// IC3Network_28767690948878
// MI455X (gfx1250) — matrix operand order repaired and checked by reading, not compiled, not run on hardware
//
#include <hip/hip_runtime.h>
#include <hip/hip_bf16.h>
#include <math.h>

// ---------------------------------------------------------------------------
// IC3Net single step on MI455X (gfx1250, wave32, WMMA).
// B=4096 A=32 S=128 H=128.  Memory-bound (~330MB @23.3TB/s ~= 14us floor):
//  - one pass per big tensor, f16 WMMA w/ f32 accumulate, fused epilogues
//  - weights pre-converted to f16 once (L2-resident), f16 frags in inner loop
//  - B-fragment loads batched per K-step so WMMAs overlap load latency
//  - K3 enc-tile staged with gfx1250 async-to-LDS ops (ASYNCcnt)
// ---------------------------------------------------------------------------

typedef __attribute__((ext_vector_type(8)))  _Float16 v8h;
typedef __attribute__((ext_vector_type(16))) _Float16 v16h;
typedef __attribute__((ext_vector_type(8)))  float    v8f;

constexpr int BN = 4096;   // batch
constexpr int AG = 32;     // agents (power of 2 -> r = b*AG + a, a = r & 31)
constexpr int SD = 128;    // obs dim
constexpr int HD = 128;    // hidden dim
constexpr int G4 = 4 * HD; // 512 gate width
constexpr int KP = 136;    // LDS row pitch (f16 elems), multiple of 8, bank-skewed

__device__ __forceinline__ v16h cat8(v8h lo, v8h hi) {
  return __builtin_shufflevector(lo, hi, 0,1,2,3,4,5,6,7,8,9,10,11,12,13,14,15);
}

// A-matrix fragment (16x32 f16) from an LDS tile row:
// lanes 0-15 hold row M=lane, K=0..15; lanes 16-31 K=16..31 (matches B frag).
__device__ __forceinline__ v16h load_afrag(const _Float16* rowbase /* &T[row*KP+kb] */,
                                           int khalf) {
  v8h lo = *(const v8h*)(rowbase + khalf * 16);
  v8h hi = *(const v8h*)(rowbase + khalf * 16 + 8);
  return cat8(lo, hi);
}

// B-matrix fragment (32x16 f16): lane<16 = column N=lane K=0..15,
// lane>=16 = same column K=16..31 -> 16 contiguous f16 of one weight row.
__device__ __forceinline__ v16h load_bfrag_f16(const _Float16* __restrict__ p) {
  v8h lo = *(const v8h*)(p);
  v8h hi = *(const v8h*)(p + 8);
  return cat8(lo, hi);
}

__device__ __forceinline__ float sigf(float x) { return 1.0f / (1.0f + __expf(-x)); }

// ---------------------------------------------------------------------------
// K0: f32 -> f16 bulk conversion (weights; done once, ~8MB -> 4MB, L2-hot)
// ---------------------------------------------------------------------------
__global__ void cvt_f16(const float* __restrict__ src, _Float16* __restrict__ dst,
                        int n8) {
  int i = blockIdx.x * blockDim.x + threadIdx.x;
  if (i < n8) {
    float4 a = *(const float4*)(src + (size_t)i * 8);
    float4 b = *(const float4*)(src + (size_t)i * 8 + 4);
    v8h r;
    r[0]=(_Float16)a.x; r[1]=(_Float16)a.y; r[2]=(_Float16)a.z; r[3]=(_Float16)a.w;
    r[4]=(_Float16)b.x; r[5]=(_Float16)b.y; r[6]=(_Float16)b.z; r[7]=(_Float16)b.w;
    *(v8h*)(dst + (size_t)i * 8) = r;
  }
}

// ---------------------------------------------------------------------------
// K1: hh[a,g] = h0[a,:].W_hh[a,g,:] + b_ih[a,g] + b_hh[a,g]   (tiny: 2M MACs)
//     also zero the per-agent argmax keys (used by K3's atomicMax).
// ---------------------------------------------------------------------------
__global__ void hh_precomp(const float* __restrict__ h0, const float* __restrict__ W_hh,
                           const float* __restrict__ b_ih, const float* __restrict__ b_hh,
                           float* __restrict__ hh, unsigned long long* __restrict__ amax) {
  const int a = blockIdx.x, t = threadIdx.x;
  if (blockIdx.x == 0 && t < AG) amax[t] = 0ull;
  for (int g = t; g < G4; g += blockDim.x) {
    const float* w = &W_hh[((size_t)a * G4 + g) * HD];
    const float* h = &h0[a * HD];
    float s = 0.f;
    #pragma unroll 8
    for (int k = 0; k < HD; ++k) s += h[k] * w[k];
    hh[a * G4 + g] = s + b_ih[a * G4 + g] + b_hh[a * G4 + g];
  }
}

// ---------------------------------------------------------------------------
// K2: enc[r,h] = sum_s obs[r,s]*We[h,s] + be[h] + comm[r,h], stored as f16.
//     M = B*A = 131072 rows; 128-row tile per block, 16 rows per wave,
//     each wave: 8 N-tiles x 4 K-steps of v_wmma_f32_16x16x32_f16.
// ---------------------------------------------------------------------------
__global__ void enc_gemm(const float* __restrict__ obs, const float* __restrict__ comm,
                         const _Float16* __restrict__ WeH, const float* __restrict__ be,
                         _Float16* __restrict__ encH) {
  __shared__ __attribute__((aligned(16))) _Float16 As[128 * KP];
  const int t = threadIdx.x;
  const int rBase = blockIdx.x * 128;

  // stage 128x128 obs tile f32->f16 (16 float4 per thread)
  #pragma unroll
  for (int i = 0; i < 16; ++i) {
    int fi = t + i * 256;
    int row = fi >> 5, c4 = (fi & 31) * 4;
    float4 o = *(const float4*)&obs[(size_t)(rBase + row) * SD + c4];
    As[row * KP + c4 + 0] = (_Float16)o.x;
    As[row * KP + c4 + 1] = (_Float16)o.y;
    As[row * KP + c4 + 2] = (_Float16)o.z;
    As[row * KP + c4 + 3] = (_Float16)o.w;
  }
  __syncthreads();

  const int lane = t & 31, wave = t >> 5;
  const int rsel = lane & 15, khalf = lane >> 4;
  const int mBase = wave * 16;

  const v8f z = {0.f,0.f,0.f,0.f,0.f,0.f,0.f,0.f};
  v8f acc[8];
  #pragma unroll
  for (int j = 0; j < 8; ++j) acc[j] = z;

  #pragma unroll
  for (int kb = 0; kb < SD; kb += 32) {
    v16h af = load_afrag(&As[(mBase + rsel) * KP + kb], khalf);
    // batch all 8 B-fragment loads first -> one long clause, staggered waits,
    // WMMAs then overlap the remaining load latency instead of waitcnt 0 each.
    v16h bf[8];
    #pragma unroll
    for (int j = 0; j < 8; ++j)
      bf[j] = load_bfrag_f16(&WeH[(size_t)(j * 16 + rsel) * SD + kb + khalf * 16]);
    #pragma unroll
    for (int j = 0; j < 8; ++j)
      acc[j] = __builtin_amdgcn_wmma_f32_16x16x32_f16(false, af, false, bf[j],
                                                      (short)0, acc[j], false, false);
  }

  // epilogue: +be +comm, f16 store. C/D layout: vgpr v -> M=v (lanes<16) / 8+v.
  const int rowOff = khalf * 8;
  #pragma unroll
  for (int j = 0; j < 8; ++j) {
    int n = j * 16 + rsel;
    float bev = be[n];
    #pragma unroll
    for (int v = 0; v < 8; ++v) {
      int r = rBase + mBase + rowOff + v;
      float val = acc[j][v] + bev + comm[(size_t)r * HD + n];
      encH[(size_t)r * HD + n] = (_Float16)val;
    }
  }
}

// ---------------------------------------------------------------------------
// K3: per-agent gate GEMM + fused LSTM cell + comm-gate argmax feed.
//     Block = (16 b-rows, agent a); wave w owns h-columns [16w,16w+16).
//     Each wave computes the 4 gate quadrants (i,f,g,o) of the SAME 16x16
//     (row,h) patch -> LSTM cell is pure per-lane VALU, no exchange needed.
//     enc tile is a pure f16 byte-copy -> gfx1250 async-to-LDS (ASYNCcnt).
// ---------------------------------------------------------------------------
__global__ void lstm_gemm(const _Float16* __restrict__ encH,
                          const _Float16* __restrict__ WihH,
                          const float* __restrict__ hh, const float* __restrict__ c0,
                          const float* __restrict__ Wg, const float* __restrict__ bg,
                          float* __restrict__ h1o, float* __restrict__ c1o,
                          unsigned long long* __restrict__ amax) {
  __shared__ __attribute__((aligned(16))) _Float16 Es[16 * KP];
  __shared__ float sd0[8][16], sd1[8][16];
  const int t = threadIdx.x;
  const int bBase = blockIdx.x * 16;
  const int a = blockIdx.y;

  // stage 16x128 enc tile: one async 16B LDS copy per thread (no VGPR data)
  {
    int row = t >> 4, col = (t & 15) * 8;
    unsigned ldsa = (unsigned)(uintptr_t)&Es[row * KP + col];
    unsigned long long ga =
        (unsigned long long)(uintptr_t)&encH[((size_t)(bBase + row) * AG + a) * HD + col];
    asm volatile("global_load_async_to_lds_b128 %0, %1, off"
                 :: "v"(ldsa), "v"(ga) : "memory");
    asm volatile("s_wait_asynccnt 0x0" ::: "memory");
  }
  __syncthreads();

  const int lane = t & 31, wave = t >> 5;
  const int rsel = lane & 15, khalf = lane >> 4;
  const int h = wave * 16 + rsel;

  // hint L2 about this agent's weight rows
  __builtin_prefetch(&WihH[((size_t)(a * G4 + h)) * HD], 0, 0);
  __builtin_prefetch(&WihH[((size_t)(a * G4 + 2 * HD + h)) * HD], 0, 0);

  const v8f z = {0.f,0.f,0.f,0.f,0.f,0.f,0.f,0.f};
  v8f acc[4];
  #pragma unroll
  for (int q = 0; q < 4; ++q) acc[q] = z;

  #pragma unroll
  for (int kb = 0; kb < HD; kb += 32) {
    v16h af = load_afrag(&Es[rsel * KP + kb], khalf);
    v16h bf[4];
    #pragma unroll
    for (int q = 0; q < 4; ++q)
      bf[q] = load_bfrag_f16(
          &WihH[((size_t)(a * G4 + q * HD + h)) * HD + kb + khalf * 16]);
    #pragma unroll
    for (int q = 0; q < 4; ++q)
      acc[q] = __builtin_amdgcn_wmma_f32_16x16x32_f16(false, af, false, bf[q],
                                                      (short)0, acc[q], false, false);
  }

  // fused LSTM cell (torch gate order i,f,g,o); h0/c0 broadcast over batch
  const float hi = hh[a * G4 + 0 * HD + h];
  const float hf = hh[a * G4 + 1 * HD + h];
  const float hg = hh[a * G4 + 2 * HD + h];
  const float ho = hh[a * G4 + 3 * HD + h];
  const float c0v = c0[a * HD + h];
  const float wg0 = Wg[(a * 2 + 0) * HD + h];
  const float wg1 = Wg[(a * 2 + 1) * HD + h];
  const int rowOff = khalf * 8;

  #pragma unroll
  for (int v = 0; v < 8; ++v) {
    const int m = rowOff + v;
    const int b = bBase + m;
    float ig = acc[0][v] + hi, fg = acc[1][v] + hf;
    float gg = acc[2][v] + hg, og = acc[3][v] + ho;
    float c1 = sigf(fg) * c0v + sigf(ig) * tanhf(gg);
    float h1 = sigf(og) * tanhf(c1);
    size_t off = ((size_t)b * AG + a) * HD + h;
    h1o[off] = h1;
    c1o[off] = c1;
    // deterministic partial dots for the 2-way comm gate (reduce 16 h-cols)
    float p0 = h1 * wg0, p1 = h1 * wg1;
    #pragma unroll
    for (int msk = 1; msk < 16; msk <<= 1) {
      p0 += __shfl_xor(p0, msk, 32);
      p1 += __shfl_xor(p1, msk, 32);
    }
    if (rsel == 0) { sd0[wave][m] = p0; sd1[wave][m] = p1; }
  }
  __syncthreads();

  if (t < 16) {  // one thread per b-row: 2-way softmax + packed argmax key
    float l0 = bg[a * 2 + 0], l1 = bg[a * 2 + 1];
    #pragma unroll
    for (int w = 0; w < 8; ++w) { l0 += sd0[w][t]; l1 += sd1[w][t]; }
    float mx = fmaxf(l0, l1);
    float e0 = __expf(l0 - mx), e1 = __expf(l1 - mx);
    float p0 = e0 / (e0 + e1), p1 = e1 / (e0 + e1);
    int action = (p1 > p0) ? 1 : 0;          // tie -> action 0 (argmax-first)
    float pb = action ? p1 : p0;
    unsigned int fidx = (unsigned)(bBase + t) * 2u + (unsigned)action;
    // key: prob bits (positive floats order-preserve as uints) in hi32,
    // ~flat_index in lo32 so ties pick the SMALLEST flat index (jnp.argmax).
    unsigned long long key = ((unsigned long long)__float_as_uint(pb) << 32) |
                             (unsigned long long)(0xFFFFFFFFu - fidx);
    atomicMax(&amax[a], key);
  }
}

// ---------------------------------------------------------------------------
// K4a: decode per-agent flattened argmax index -> float multiplier
// ---------------------------------------------------------------------------
__global__ void decode_idx(const unsigned long long* __restrict__ amax,
                           float* __restrict__ idxf) {
  int a = threadIdx.x;
  if (a < AG)
    idxf[a] = (float)(0xFFFFFFFFu - (unsigned)(amax[a] & 0xFFFFFFFFull));
}

// ---------------------------------------------------------------------------
// K4b: comm_sum[b,h] = norm * sum_a idx[a]*h1[b,a,h]
// ---------------------------------------------------------------------------
__global__ void comm_reduce(const float* __restrict__ h1o, const float* __restrict__ idxf,
                            const int* __restrict__ alive, float* __restrict__ csum) {
  __shared__ float sidx[AG];
  const int t = threadIdx.x;
  if (t < AG) sidx[t] = idxf[t];
  __syncthreads();
  const int id = blockIdx.x * blockDim.x + t;  // 0..B*H-1
  const int b = id >> 7, h = id & (HD - 1);
  const float* base = &h1o[(size_t)b * AG * HD + h];
  float s = 0.f;
  #pragma unroll
  for (int a = 0; a < AG; ++a) s += base[a * HD] * sidx[a];
  const int al = alive[0];
  const float norm = al ? 1.0f / (float)(al - 1) : 1.0f;
  csum[id] = s * norm;
}

// ---------------------------------------------------------------------------
// K5: out[r,:] = softmax_o( (csum[b,:] - idx[a]*h1[r,:]) . Wt[o,:] + bt[o] )
//     Same tiling as K2; row-softmax fused via in-lane + shfl_xor(1,2,4,8)
//     reductions over the 16-lane C/D halves.
// ---------------------------------------------------------------------------
__global__ void out_gemm(const float* __restrict__ h1o, const float* __restrict__ csum,
                         const float* __restrict__ idxf, const _Float16* __restrict__ WtH,
                         const float* __restrict__ bt, float* __restrict__ out) {
  __shared__ __attribute__((aligned(16))) _Float16 Xs[128 * KP];
  __shared__ float sidx[AG];
  const int t = threadIdx.x;
  const int rBase = blockIdx.x * 128;
  if (t < AG) sidx[t] = idxf[t];
  __syncthreads();

  // stage X[r,h] = csum[b,h] - idx[a]*h1[r,h] as f16 (r = b*32 + a)
  #pragma unroll
  for (int i = 0; i < 16; ++i) {
    int fi = t + i * 256;
    int row = fi >> 5, c4 = (fi & 31) * 4;
    int r = rBase + row;
    float sc = sidx[r & (AG - 1)];
    int b = r >> 5;
    float4 hv = *(const float4*)&h1o[(size_t)r * HD + c4];
    float4 cv = *(const float4*)&csum[(size_t)b * HD + c4];
    Xs[row * KP + c4 + 0] = (_Float16)(cv.x - sc * hv.x);
    Xs[row * KP + c4 + 1] = (_Float16)(cv.y - sc * hv.y);
    Xs[row * KP + c4 + 2] = (_Float16)(cv.z - sc * hv.z);
    Xs[row * KP + c4 + 3] = (_Float16)(cv.w - sc * hv.w);
  }
  __syncthreads();

  const int lane = t & 31, wave = t >> 5;
  const int rsel = lane & 15, khalf = lane >> 4;
  const int mBase = wave * 16;

  const v8f z = {0.f,0.f,0.f,0.f,0.f,0.f,0.f,0.f};
  v8f acc[8];
  #pragma unroll
  for (int j = 0; j < 8; ++j) acc[j] = z;

  #pragma unroll
  for (int kb = 0; kb < HD; kb += 32) {
    v16h af = load_afrag(&Xs[(mBase + rsel) * KP + kb], khalf);
    v16h bf[8];
    #pragma unroll
    for (int j = 0; j < 8; ++j)
      bf[j] = load_bfrag_f16(&WtH[(size_t)(j * 16 + rsel) * HD + kb + khalf * 16]);
    #pragma unroll
    for (int j = 0; j < 8; ++j)
      acc[j] = __builtin_amdgcn_wmma_f32_16x16x32_f16(false, af, false, bf[j],
                                                      (short)0, acc[j], false, false);
  }

  // fused row softmax over 128 logits per row
  const int rowOff = khalf * 8;
  float btv[8];
  #pragma unroll
  for (int j = 0; j < 8; ++j) btv[j] = bt[j * 16 + rsel];

  #pragma unroll
  for (int v = 0; v < 8; ++v) {
    float e[8], mx = -3.0e38f;
    #pragma unroll
    for (int j = 0; j < 8; ++j) { e[j] = acc[j][v] + btv[j]; mx = fmaxf(mx, e[j]); }
    #pragma unroll
    for (int msk = 1; msk < 16; msk <<= 1) mx = fmaxf(mx, __shfl_xor(mx, msk, 32));
    float s = 0.f;
    #pragma unroll
    for (int j = 0; j < 8; ++j) { e[j] = __expf(e[j] - mx); s += e[j]; }
    #pragma unroll
    for (int msk = 1; msk < 16; msk <<= 1) s += __shfl_xor(s, msk, 32);
    const float inv = 1.0f / s;
    const int r = rBase + mBase + rowOff + v;
    #pragma unroll
    for (int j = 0; j < 8; ++j) out[(size_t)r * HD + j * 16 + rsel] = e[j] * inv;
  }
}

// ---------------------------------------------------------------------------
// launch
// ---------------------------------------------------------------------------
extern "C" void kernel_launch(void* const* d_in, const int* in_sizes, int n_in,
                              void* d_out, int out_size, void* d_ws, size_t ws_size,
                              hipStream_t stream) {
  (void)in_sizes; (void)n_in; (void)out_size; (void)ws_size;

  const float* obs   = (const float*)d_in[0];
  const float* comm  = (const float*)d_in[1];
  const float* h0    = (const float*)d_in[2];
  const float* c0    = (const float*)d_in[3];
  const float* We    = (const float*)d_in[4];
  const float* be    = (const float*)d_in[5];
  const float* W_ih  = (const float*)d_in[6];
  const float* W_hh  = (const float*)d_in[7];
  const float* b_ih  = (const float*)d_in[8];
  const float* b_hh  = (const float*)d_in[9];
  const float* Wg    = (const float*)d_in[10];
  const float* bg    = (const float*)d_in[11];
  const float* Wt    = (const float*)d_in[12];
  const float* bt    = (const float*)d_in[13];
  const int*   alive = (const int*)d_in[14];

  const size_t N = (size_t)BN * AG * HD;  // 16,777,216
  float* out = (float*)d_out;             // [B,A,H]
  float* h1o = out + N;                   // [B,A,H]
  float* c1o = h1o + N;                   // [B,A,H]

  // workspace layout
  char* ws = (char*)d_ws;
  _Float16*           encH = (_Float16*)ws;                        // 32 MiB
  char* p = ws + N * sizeof(_Float16);
  float*              csum = (float*)p;  p += (size_t)BN * HD * 4; //  2 MiB
  float*              hh   = (float*)p;  p += (size_t)AG * G4 * 4; // 64 KiB
  unsigned long long* amax = (unsigned long long*)p; p += AG * 8;
  float*              idxf = (float*)p;  p += AG * 4;
  p = (char*)(((uintptr_t)p + 255) & ~(uintptr_t)255);
  _Float16*           WeH  = (_Float16*)p; p += (size_t)HD * SD * 2;       // 32 KiB
  _Float16*           WtH  = (_Float16*)p; p += (size_t)HD * HD * 2;       // 32 KiB
  _Float16*           WihH = (_Float16*)p;                                 //  4 MiB

  // weight pre-conversion (once per call; tiny)
  cvt_f16<<<(HD * SD / 8 + 255) / 256, 256, 0, stream>>>(We, WeH, HD * SD / 8);
  cvt_f16<<<(HD * HD / 8 + 255) / 256, 256, 0, stream>>>(Wt, WtH, HD * HD / 8);
  cvt_f16<<<(AG * G4 * HD / 8 + 255) / 256, 256, 0, stream>>>(W_ih, WihH,
                                                              AG * G4 * HD / 8);

  hh_precomp <<<AG, 256, 0, stream>>>(h0, W_hh, b_ih, b_hh, hh, amax);
  enc_gemm   <<<(BN * AG) / 128, 256, 0, stream>>>(obs, comm, WeH, be, encH);
  lstm_gemm  <<<dim3(BN / 16, AG), 256, 0, stream>>>(encH, WihH, hh, c0, Wg, bg,
                                                     h1o, c1o, amax);
  decode_idx <<<1, 32, 0, stream>>>(amax, idxf);
  comm_reduce<<<(BN * HD) / 256, 256, 0, stream>>>(h1o, idxf, alive, csum);
  out_gemm   <<<(BN * AG) / 128, 256, 0, stream>>>(h1o, csum, idxf, WtH, bt, out);
}
